// SetAbstraction_28570122453861
// MI455X (gfx1250) — matrix operand order repaired and checked by reading, not compiled, not run on hardware
//
#include <hip/hip_runtime.h>
#include <hip/hip_bf16.h>

typedef __attribute__((ext_vector_type(16))) _Float16 v16h;
typedef __attribute__((ext_vector_type(8)))  _Float16 v8h;
typedef __attribute__((ext_vector_type(8)))  float    v8f;

#define B_   16
#define N_   4096
#define CIN  64
#define M_   1024
#define K_   32
#define R2_  0.04f      // R=0.2
#define EPS_ 1e-5f
#define KP1  96         // layer-1 K padded 67 -> 96

// ---------------------------------------------------------------------------
// FPS: one block per batch; points live in VGPRs; 2 barriers per iteration.
// ---------------------------------------------------------------------------
#define FPS_T   256
#define FPS_PER (N_ / FPS_T)   // 16 points per thread (blocked for tie-break order)

__global__ __launch_bounds__(FPS_T) void fps_kernel(const float* __restrict__ pos,
                                                    float* __restrict__ cpos)
{
    const int b = blockIdx.x;
    const int t = threadIdx.x;
    const float* pb = pos  + (size_t)b * N_ * 3;
    float*       cp = cpos + (size_t)b * M_ * 3;

    __shared__ float rv[FPS_T / 32];
    __shared__ int   ri[FPS_T / 32];
    __shared__ float lastP[3];

    float px[FPS_PER], py[FPS_PER], pz[FPS_PER], mind[FPS_PER];
    const int i0 = t * FPS_PER;
#pragma unroll
    for (int j = 0; j < FPS_PER; ++j) {
        px[j] = pb[(i0 + j) * 3 + 0];
        py[j] = pb[(i0 + j) * 3 + 1];
        pz[j] = pb[(i0 + j) * 3 + 2];
        mind[j] = 1e10f;
    }
    if (t == 0) {
        lastP[0] = pb[0]; lastP[1] = pb[1]; lastP[2] = pb[2];
        cp[0] = pb[0]; cp[1] = pb[1]; cp[2] = pb[2];   // idx[0] = 0
    }
    __syncthreads();

    const int lane = t & 31, wave = t >> 5;
    for (int it = 1; it < M_; ++it) {
        const float lx = lastP[0], ly = lastP[1], lz = lastP[2];
        float bv = -1.0f; int bi = 0;
#pragma unroll
        for (int j = 0; j < FPS_PER; ++j) {
            const float dx = px[j] - lx, dy = py[j] - ly, dz = pz[j] - lz;
            const float d  = dx * dx + dy * dy + dz * dz;
            const float nm = fminf(mind[j], d);
            mind[j] = nm;
            if (nm > bv) { bv = nm; bi = i0 + j; }   // strict > keeps first index
        }
        // wave32 butterfly argmax (value desc, index asc on ties)
#pragma unroll
        for (int m = 16; m > 0; m >>= 1) {
            const float ov = __shfl_xor(bv, m, 32);
            const int   oi = __shfl_xor(bi, m, 32);
            if (ov > bv || (ov == bv && oi < bi)) { bv = ov; bi = oi; }
        }
        if (lane == 0) { rv[wave] = bv; ri[wave] = bi; }
        __syncthreads();
        if (t == 0) {
            float fv = rv[0]; int fi = ri[0];
#pragma unroll
            for (int w = 1; w < FPS_T / 32; ++w) {
                const float ov = rv[w]; const int oi = ri[w];
                if (ov > fv || (ov == fv && oi < fi)) { fv = ov; fi = oi; }
            }
            const float sx = pb[fi * 3 + 0], sy = pb[fi * 3 + 1], sz = pb[fi * 3 + 2];
            lastP[0] = sx; lastP[1] = sy; lastP[2] = sz;
            cp[it * 3 + 0] = sx; cp[it * 3 + 1] = sy; cp[it * 3 + 2] = sz;
        }
        __syncthreads();
    }
}

// ---------------------------------------------------------------------------
// Fragment loaders (LDS -> VGPR) following CDNA5 16-bit WMMA layouts.
// ---------------------------------------------------------------------------
union F16x16 { v16h v; v8h h[2]; };

// A (16x32 f16): lane holds row M=lane%16; VGPR0-3 = K {0..7|8..15}, VGPR4-7 = +16
__device__ inline v16h load_a_frag(const _Float16* aB, int t, int s, int lane, int stride)
{
    const int row = 16 * t + (lane & 15);
    const int hs  = lane >> 4;
    const _Float16* p = aB + row * stride + s * 32 + 8 * hs;
    F16x16 u;
    u.h[0] = *(const v8h*)(p);
    u.h[1] = *(const v8h*)(p + 16);
    return u.v;
}

// B (32x16 f16), weights stored transposed [n][k]: lane holds col n=lane%16,
// K {0..7|8..15} by lane/16, then +16 (matches A layout).
__device__ inline v16h load_b_frag(const _Float16* wT, int kpw, int cBase, int s, int lane)
{
    const int n  = cBase + (lane & 15);
    const int hs = lane >> 4;
    const _Float16* p = wT + n * kpw + s * 32 + 8 * hs;
    F16x16 u;
    u.h[0] = *(const v8h*)(p);
    u.h[1] = *(const v8h*)(p + 16);
    return u.v;
}

// ---------------------------------------------------------------------------
// Fused ball-query + gather + 3-layer WMMA MLP + masked max. 1 wave / centroid.
// ---------------------------------------------------------------------------
#define WAVES 4
#define MLP_T (WAVES * 32)

__global__ __launch_bounds__(MLP_T) void sa_mlp_kernel(
    const float* __restrict__ x,   const float* __restrict__ pos,
    const float* __restrict__ W1,  const float* __restrict__ g1, const float* __restrict__ b1,
    const float* __restrict__ rm1, const float* __restrict__ rv1,
    const float* __restrict__ W2,  const float* __restrict__ g2, const float* __restrict__ b2,
    const float* __restrict__ rm2, const float* __restrict__ rv2,
    const float* __restrict__ W3,  const float* __restrict__ g3, const float* __restrict__ b3,
    const float* __restrict__ rm3, const float* __restrict__ rv3,
    const float* __restrict__ cpos, float* __restrict__ xout)
{
    __shared__ alignas(16) _Float16 wT1[64 * KP1];   // [n][k] BN-folded, K padded
    __shared__ alignas(16) _Float16 wT2[64 * 64];
    __shared__ alignas(16) _Float16 wT3[128 * 64];
    __shared__ float bb1[64], bb2[64], bb3[128];
    __shared__ alignas(16) _Float16 aBuf[WAVES][32 * KP1];
    __shared__ int nbr[WAVES][K_];

    const int tid = threadIdx.x;

    // ---- fold BN into f16 transposed weights (per block, tiny) ----
    for (int i = tid; i < 64 * KP1; i += MLP_T) {
        const int n = i / KP1, k = i % KP1;
        const float s = g1[n] * rsqrtf(rv1[n] + EPS_);
        wT1[i] = (k < 67) ? (_Float16)(W1[k * 64 + n] * s) : (_Float16)0.f;
    }
    for (int i = tid; i < 64 * 64; i += MLP_T) {
        const int n = i / 64, k = i % 64;
        const float s = g2[n] * rsqrtf(rv2[n] + EPS_);
        wT2[i] = (_Float16)(W2[k * 64 + n] * s);
    }
    for (int i = tid; i < 128 * 64; i += MLP_T) {
        const int n = i / 64, k = i % 64;
        const float s = g3[n] * rsqrtf(rv3[n] + EPS_);
        wT3[i] = (_Float16)(W3[k * 128 + n] * s);
    }
    for (int n = tid; n < 64; n += MLP_T) {
        const float s1 = g1[n] * rsqrtf(rv1[n] + EPS_);
        const float s2 = g2[n] * rsqrtf(rv2[n] + EPS_);
        bb1[n] = b1[n] - rm1[n] * s1;
        bb2[n] = b2[n] - rm2[n] * s2;
    }
    for (int n = tid; n < 128; n += MLP_T) {
        const float s3 = g3[n] * rsqrtf(rv3[n] + EPS_);
        bb3[n] = b3[n] - rm3[n] * s3;
    }

    const int wave = tid >> 5, lane = tid & 31;
    const int gcent = blockIdx.x * WAVES + wave;
    const int b = gcent / M_, cent = gcent % M_;
    const float* pb = pos  + (size_t)b * N_ * 3;
    const float* cp = cpos + ((size_t)b * M_ + cent) * 3;
    const float cx = cp[0], cy = cp[1], cz = cp[2];

    // ---- ball query: first K in-radius points in index order ----
    int* nb = nbr[wave];
    unsigned validMask;
    {
        int cnt = 0;
        for (int base = 0; base < N_ && cnt < K_; base += 32) {
            const int j = base + lane;
            const float dx = pb[j * 3 + 0] - cx;
            const float dy = pb[j * 3 + 1] - cy;
            const float dz = pb[j * 3 + 2] - cz;
            const bool in = (dx * dx + dy * dy + dz * dz) <= R2_;
            const unsigned mask = (unsigned)__ballot(in);
            const int pre = __popc(mask & ((1u << lane) - 1u));
            if (in && (cnt + pre) < K_) nb[cnt + pre] = j;
            cnt += __popc(mask);
            if (cnt > K_) cnt = K_;
        }
        __builtin_amdgcn_wave_barrier();
        if (lane >= cnt) nb[lane] = 0;                    // reference pads with index 0
        validMask = (cnt >= 32) ? 0xffffffffu : ((1u << cnt) - 1u);
    }
    __builtin_amdgcn_wave_barrier();

    // ---- gather [x_j | pos_j - cpos] -> f16 tile, lane = neighbor slot ----
    _Float16* aB = &aBuf[wave][0];
    {
        const int nid = nb[lane];
        const float* xr = x + ((size_t)b * N_ + nid) * CIN;
        _Float16* row = aB + lane * KP1;
#pragma unroll 8
        for (int c = 0; c < CIN; ++c) row[c] = (_Float16)xr[c];
        row[64] = (_Float16)(pb[nid * 3 + 0] - cx);
        row[65] = (_Float16)(pb[nid * 3 + 1] - cy);
        row[66] = (_Float16)(pb[nid * 3 + 2] - cz);
#pragma unroll
        for (int c = 67; c < KP1; ++c) row[c] = (_Float16)0.f;
    }
    __syncthreads();   // weights ready + tiles ready

    const int hs = lane >> 4;
    const int nlo = lane & 15;

    // ---- layer 1: [32x96] @ [96x64] ----
    v16h a1[2][3];
#pragma unroll
    for (int t = 0; t < 2; ++t)
#pragma unroll
        for (int s = 0; s < 3; ++s) a1[t][s] = load_a_frag(aB, t, s, lane, KP1);
#pragma unroll
    for (int c = 0; c < 4; ++c) {
        const float bias = bb1[c * 16 + nlo];
#pragma unroll
        for (int t = 0; t < 2; ++t) {
            v8f acc;
#pragma unroll
            for (int r = 0; r < 8; ++r) acc[r] = bias;
#pragma unroll
            for (int s = 0; s < 3; ++s)
                acc = __builtin_amdgcn_wmma_f32_16x16x32_f16(
                        false, a1[t][s], false, load_b_frag(wT1, KP1, c * 16, s, lane),
                        (short)0, acc, false, false);
#pragma unroll
            for (int r = 0; r < 8; ++r)   // ReLU + store back (inputs already in regs)
                aB[(16 * t + r + 8 * hs) * KP1 + c * 16 + nlo] = (_Float16)fmaxf(acc[r], 0.f);
        }
    }
    __builtin_amdgcn_wave_barrier();

    // ---- layer 2: [32x64] @ [64x64] ----
    v16h a2[2][2];
#pragma unroll
    for (int t = 0; t < 2; ++t)
#pragma unroll
        for (int s = 0; s < 2; ++s) a2[t][s] = load_a_frag(aB, t, s, lane, KP1);
#pragma unroll
    for (int c = 0; c < 4; ++c) {
        const float bias = bb2[c * 16 + nlo];
#pragma unroll
        for (int t = 0; t < 2; ++t) {
            v8f acc;
#pragma unroll
            for (int r = 0; r < 8; ++r) acc[r] = bias;
#pragma unroll
            for (int s = 0; s < 2; ++s)
                acc = __builtin_amdgcn_wmma_f32_16x16x32_f16(
                        false, a2[t][s], false, load_b_frag(wT2, 64, c * 16, s, lane),
                        (short)0, acc, false, false);
#pragma unroll
            for (int r = 0; r < 8; ++r)
                aB[(16 * t + r + 8 * hs) * KP1 + c * 16 + nlo] = (_Float16)fmaxf(acc[r], 0.f);
        }
    }
    __builtin_amdgcn_wave_barrier();

    // ---- layer 3: [32x64] @ [64x128] fused with masked max over neighbors ----
    v16h a3[2][2];
#pragma unroll
    for (int t = 0; t < 2; ++t)
#pragma unroll
        for (int s = 0; s < 2; ++s) a3[t][s] = load_a_frag(aB, t, s, lane, KP1);
    float* xr = xout + ((size_t)b * M_ + cent) * 128;
#pragma unroll
    for (int c = 0; c < 8; ++c) {
        const float bias = bb3[c * 16 + nlo];
        float best = -__builtin_inff();
#pragma unroll
        for (int t = 0; t < 2; ++t) {
            v8f acc;
#pragma unroll
            for (int r = 0; r < 8; ++r) acc[r] = bias;
#pragma unroll
            for (int s = 0; s < 2; ++s)
                acc = __builtin_amdgcn_wmma_f32_16x16x32_f16(
                        false, a3[t][s], false, load_b_frag(wT3, 64, c * 16, s, lane),
                        (short)0, acc, false, false);
#pragma unroll
            for (int r = 0; r < 8; ++r) {
                const int nbi = 16 * t + r + 8 * hs;     // neighbor (row) index of acc[r]
                if ((validMask >> nbi) & 1u) best = fmaxf(best, fmaxf(acc[r], 0.f));
            }
        }
        best = fmaxf(best, __shfl_xor(best, 16, 32));    // merge lane halves (same column)
        if (lane < 16) xr[c * 16 + lane] = best;
    }
}

// ---------------------------------------------------------------------------
extern "C" void kernel_launch(void* const* d_in, const int* in_sizes, int n_in,
                              void* d_out, int out_size, void* d_ws, size_t ws_size,
                              hipStream_t stream)
{
    (void)in_sizes; (void)n_in; (void)out_size; (void)d_ws; (void)ws_size;
    const float* x   = (const float*)d_in[0];
    const float* pos = (const float*)d_in[1];
    const float* W1  = (const float*)d_in[2];
    const float* g1  = (const float*)d_in[3];
    const float* b1  = (const float*)d_in[4];
    const float* rm1 = (const float*)d_in[5];
    const float* rv1 = (const float*)d_in[6];
    const float* W2  = (const float*)d_in[7];
    const float* g2  = (const float*)d_in[8];
    const float* b2  = (const float*)d_in[9];
    const float* rm2 = (const float*)d_in[10];
    const float* rv2 = (const float*)d_in[11];
    const float* W3  = (const float*)d_in[12];
    const float* g3  = (const float*)d_in[13];
    const float* b3  = (const float*)d_in[14];
    const float* rm3 = (const float*)d_in[15];
    const float* rv3 = (const float*)d_in[16];

    float* xout    = (float*)d_out;                              // [B,M,128]
    float* cposOut = (float*)d_out + (size_t)B_ * M_ * 128;      // [B,M,3]

    fps_kernel<<<B_, FPS_T, 0, stream>>>(pos, cposOut);

    const int nBlocks = (B_ * M_) / WAVES;                       // 4096
    sa_mlp_kernel<<<nBlocks, MLP_T, 0, stream>>>(
        x, pos,
        W1, g1, b1, rm1, rv1,
        W2, g2, b2, rm2, rv2,
        W3, g3, b3, rm3, rv3,
        cposOut, xout);
}
